// EdgeConvAuxLayer_23905787970103
// MI455X (gfx1250) — matrix operand order repaired and checked by reading, not compiled, not run on hardware
//
#include <hip/hip_runtime.h>
#include <hip/hip_bf16.h>

typedef __attribute__((ext_vector_type(16))) _Float16 v16h;
typedef __attribute__((ext_vector_type(8)))  _Float16 v8h;
typedef __attribute__((ext_vector_type(8)))  float    v8f;

#define P_PTS   65536
#define KNB     16
#define E_TOT   (P_PTS * KNB)
#define OUTF    64
#define EPSV    1e-5f

// workspace layout (floats)
#define WS_S1     0      // 27 accumulators: mean(6) + upper-tri second moments(21)
#define WS_W1T    32     // 64x32 f32, BN1-folded W1, transposed (n-major, K padded 6->32)
#define WS_B1HAT  2080   // 64
#define WS_S2SUM  2144   // 64
#define WS_S2SQ   2208   // 64
#define WS_SCALE2 2272   // 64
#define WS_SHIFT2 2336   // 64
#define WS_TOTAL  2400

// ---------------------------------------------------------------- WMMA helpers

__device__ __forceinline__ v8f wmma16(v16h a, v16h b, v8f c) {
  return __builtin_amdgcn_wmma_f32_16x16x32_f16(false, a, false, b, (short)0, c,
                                                false, false);
}

// A-operand (16x32 f16, M rows striped per lane): lane<16 -> row=lane, halves
// k 0..7 and 16..23; lane>=16 -> row=lane-16, halves k 8..15 and 24..31.
__device__ __forceinline__ v16h ld_fragA(const _Float16* base, int stride) {
  const int lane = threadIdx.x & 31;
  const int r    = lane & 15;
  const bool hi  = lane >= 16;
  const _Float16* q = base + r * stride;
  v8h lo = *reinterpret_cast<const v8h*>(q + (hi ? 8  : 0));
  v8h up = *reinterpret_cast<const v8h*>(q + (hi ? 24 : 16));
  v16h v;
#pragma unroll
  for (int i = 0; i < 8; ++i) { v[i] = lo[i]; v[8 + i] = up[i]; }
  return v;
}

// B-operand (32x16 f16, N cols striped per lane): lane<16 -> col=lane, K 0..7
// and 16..23; lane>=16 -> col=lane-16, K 8..15 and 24..31. Weights are staged
// N-major in LDS so the K-slice is contiguous.
__device__ __forceinline__ v16h ld_fragB(const _Float16* base, int stride) {
  const int lane = threadIdx.x & 31;
  const int n    = lane & 15;
  const bool hi  = lane >= 16;
  const _Float16* q = base + n * stride;
  v8h lo = *reinterpret_cast<const v8h*>(q + (hi ? 8 : 0));
  v8h up = *reinterpret_cast<const v8h*>(q + (hi ? 24 : 16));
  v16h v;
#pragma unroll
  for (int i = 0; i < 8; ++i) { v[i] = lo[i]; v[8 + i] = up[i]; }
  return v;
}

// C fragment (16x16 f32: lane = col (dup in halves), vgpr i = row (+8 for hi
// lanes)) -> relu(x+bias) -> row-major f16 staging tile.
__device__ __forceinline__ void store_rows_relu(v8f c, _Float16* stage,
                                                int stride, int colbase,
                                                float bias) {
  const int lane = threadIdx.x & 31;
  const int col  = colbase + (lane & 15);
  const int r0   = (lane & 16) ? 8 : 0;
#pragma unroll
  for (int i = 0; i < 8; ++i) {
    float v = c[i] + bias;
    v = v > 0.f ? v : 0.f;
    stage[(r0 + i) * stride + col] = (_Float16)v;
  }
}

// ---------------------------------------------------------------- gathers

__device__ __forceinline__ void gather_eg(const float* __restrict__ geom,
                                          const int* __restrict__ idxj, int p,
                                          _Float16* eg /*16x32*/) {
  const int lane = threadIdx.x & 31;
  for (int k = lane; k < 16 * 32; k += 32) eg[k] = (_Float16)0.f;
  const int r = lane & 15;
  const int j = idxj[p * KNB + r];
  if (lane < 16) {
    float xi0 = geom[p * 3 + 0], xi1 = geom[p * 3 + 1], xi2 = geom[p * 3 + 2];
    float xj0 = geom[j * 3 + 0], xj1 = geom[j * 3 + 1], xj2 = geom[j * 3 + 2];
    _Float16* row = eg + r * 32;
    row[0] = (_Float16)xi0;          row[1] = (_Float16)xi1;
    row[2] = (_Float16)xi2;          row[3] = (_Float16)(xj0 - xi0);
    row[4] = (_Float16)(xj1 - xi1);  row[5] = (_Float16)(xj2 - xi2);
  }
}

__device__ __forceinline__ void gather_ax(const float* __restrict__ aux,
                                          const int* __restrict__ idxj, int p,
                                          _Float16* ax /*16x32*/) {
  const int lane = threadIdx.x & 31;
  const int r    = lane & 15;
  const int j    = idxj[p * KNB + r];
  const float* src = (lane < 16) ? (aux + (size_t)p * 16) : (aux + (size_t)j * 16);
  _Float16* row = ax + r * 32 + ((lane < 16) ? 0 : 16);
#pragma unroll
  for (int c = 0; c < 16; ++c) row[c] = (_Float16)src[c];
}

// ---------------------------------------------------------------- stage 1: cov

__global__ __launch_bounds__(256) void s1_cov_kernel(
    const float* __restrict__ geom, const int* __restrict__ idxj,
    float* __restrict__ S) {
  float a[27];
#pragma unroll
  for (int k = 0; k < 27; ++k) a[k] = 0.f;
  const int stride = gridDim.x * blockDim.x;
  for (int e = blockIdx.x * blockDim.x + threadIdx.x; e < E_TOT; e += stride) {
    const int i = e >> 4;
    const int j = idxj[e];
    float v[6];
    v[0] = geom[i * 3 + 0]; v[1] = geom[i * 3 + 1]; v[2] = geom[i * 3 + 2];
    v[3] = geom[j * 3 + 0] - v[0];
    v[4] = geom[j * 3 + 1] - v[1];
    v[5] = geom[j * 3 + 2] - v[2];
    int idx = 6;
#pragma unroll
    for (int p = 0; p < 6; ++p) {
      a[p] += v[p];
#pragma unroll
      for (int q = p; q < 6; ++q) a[idx++] += v[p] * v[q];
    }
  }
  __shared__ float red[27];
  if (threadIdx.x < 27) red[threadIdx.x] = 0.f;
  __syncthreads();
#pragma unroll
  for (int k = 0; k < 27; ++k) atomicAdd(&red[k], a[k]);
  __syncthreads();
  if (threadIdx.x < 27) atomicAdd(&S[threadIdx.x], red[threadIdx.x]);
}

// -------------------------------------------------- fold BN1 into W1', b1'

__global__ void fold_bn1_kernel(const float* __restrict__ W1,
                                const float* __restrict__ b1,
                                const float* __restrict__ g1,
                                const float* __restrict__ bb1,
                                float* __restrict__ ws) {
  const int c = threadIdx.x;
  if (c >= 64) return;
  const float invE = 1.f / (float)E_TOT;
  float m[6];
#pragma unroll
  for (int a = 0; a < 6; ++a) m[a] = ws[WS_S1 + a] * invE;
  float C[6][6];
  int idx = 6;
#pragma unroll
  for (int p = 0; p < 6; ++p)
#pragma unroll
    for (int q = p; q < 6; ++q) {
      float cc = ws[WS_S1 + idx++] * invE - m[p] * m[q];
      C[p][q] = cc; C[q][p] = cc;
    }
  float w[6];
#pragma unroll
  for (int a = 0; a < 6; ++a) w[a] = W1[a * 64 + c];
  float mu = b1[c], var = 0.f;
#pragma unroll
  for (int a = 0; a < 6; ++a) {
    mu += m[a] * w[a];
#pragma unroll
    for (int b = 0; b < 6; ++b) var += w[a] * w[b] * C[a][b];
  }
  const float sc = g1[c] * rsqrtf(var + EPSV);
#pragma unroll
  for (int a = 0; a < 6; ++a) ws[WS_W1T + c * 32 + a] = w[a] * sc;
#pragma unroll
  for (int a = 6; a < 32; ++a) ws[WS_W1T + c * 32 + a] = 0.f;
  ws[WS_B1HAT + c] = (b1[c] - mu) * sc + bb1[c];
}

// -------------------------------------------- stage 2: BN2 stats (WMMA pass)

__global__ __launch_bounds__(256) void s2_stats_kernel(
    const float* __restrict__ geom, const int* __restrict__ idxj,
    const float* __restrict__ W2, float* __restrict__ ws) {
  __shared__ __align__(16) _Float16 w1t[64 * 32];
  __shared__ __align__(16) _Float16 w2t[64 * 64];
  __shared__ __align__(16) _Float16 st32[8][16 * 32];
  __shared__ __align__(16) _Float16 st64[8][16 * 64];
  __shared__ float acc[128];
  const int tid = threadIdx.x;
  if (tid < 128) acc[tid] = 0.f;
  for (int idx = tid; idx < 64 * 32; idx += 256)
    w1t[idx] = (_Float16)ws[WS_W1T + idx];
  for (int idx = tid; idx < 64 * 64; idx += 256) {
    int n = idx >> 6, k = idx & 63;
    w2t[idx] = (_Float16)W2[k * 64 + n];
  }
  __syncthreads();
  const int wave = tid >> 5, lane = tid & 31;
  _Float16* eg  = st32[wave];
  _Float16* hst = st64[wave];
  for (int it = 0; it < 4; ++it) {
    const int p = (blockIdx.x * 8 + wave) * 4 + it;  // tile index == point index
    gather_eg(geom, idxj, p, eg);
    v16h aEG = ld_fragA(eg, 32);
#pragma unroll
    for (int t = 0; t < 4; ++t) {
      v16h b = ld_fragB(w1t + 16 * t * 32, 32);
      v8f z = {};
      z = wmma16(aEG, b, z);
      const int col = 16 * t + (lane & 15);
      store_rows_relu(z, hst, 64, 16 * t, ws[WS_B1HAT + col]);
    }
    v8f fC[4];
#pragma unroll
    for (int t = 0; t < 4; ++t) { v8f z = {}; fC[t] = z; }
#pragma unroll
    for (int k0 = 0; k0 < 64; k0 += 32) {
      v16h aH = ld_fragA(hst + k0, 64);
#pragma unroll
      for (int t = 0; t < 4; ++t) {
        v16h b = ld_fragB(w2t + 16 * t * 64 + k0, 64);
        fC[t] = wmma16(aH, b, fC[t]);
      }
    }
#pragma unroll
    for (int t = 0; t < 4; ++t) {
      float s = 0.f, q = 0.f;
#pragma unroll
      for (int i = 0; i < 8; ++i) { float v = fC[t][i]; s += v; q += v * v; }
      s += __shfl_xor(s, 16, 32);
      q += __shfl_xor(q, 16, 32);
      if (lane < 16) {
        atomicAdd(&acc[16 * t + lane], s);
        atomicAdd(&acc[64 + 16 * t + lane], q);
      }
    }
  }
  __syncthreads();
  if (tid < 128) atomicAdd(&ws[WS_S2SUM + tid], acc[tid]);
}

__global__ void fold_bn2_kernel(const float* __restrict__ g2,
                                const float* __restrict__ bb2,
                                float* __restrict__ ws) {
  const int c = threadIdx.x;
  if (c >= 64) return;
  const float invE = 1.f / (float)E_TOT;
  const float mu  = ws[WS_S2SUM + c] * invE;           // mean of z2 (sans b2)
  const float var = ws[WS_S2SQ + c] * invE - mu * mu;  // b2 shift cancels in var
  const float sc  = g2[c] * rsqrtf(var + EPSV);
  ws[WS_SCALE2 + c] = sc;
  ws[WS_SHIFT2 + c] = -mu * sc + bb2[c];               // b2 cancels analytically
}

// ------------------------------------- stage 3: fused main pass (32 WMMA/wave)

__global__ __launch_bounds__(256) void s3_main_kernel(
    const float* __restrict__ geom, const float* __restrict__ aux,
    const int* __restrict__ idxj, const float* __restrict__ W2,
    const float* __restrict__ Wa1, const float* __restrict__ ba1,
    const float* __restrict__ Wa2, const float* __restrict__ ba2,
    const float* __restrict__ lng, const float* __restrict__ lnb,
    const float* __restrict__ ws, float* __restrict__ out) {
  __shared__ __align__(16) _Float16 w1t[64 * 32];
  __shared__ __align__(16) _Float16 w2t[64 * 64];
  __shared__ __align__(16) _Float16 wa1t[64 * 32];
  __shared__ __align__(16) _Float16 wa2t[128 * 64];
  __shared__ __align__(16) _Float16 st32[8][16 * 32];   // reused: eg, then aux
  __shared__ __align__(16) _Float16 st64[8][16 * 64];   // reused: h, then u
  const int tid = threadIdx.x;
  for (int idx = tid; idx < 64 * 32; idx += 256) {
    w1t[idx] = (_Float16)ws[WS_W1T + idx];
    int n = idx >> 5, k = idx & 31;
    wa1t[idx] = (_Float16)Wa1[k * 64 + n];
  }
  for (int idx = tid; idx < 64 * 64; idx += 256) {
    int n = idx >> 6, k = idx & 63;
    w2t[idx] = (_Float16)W2[k * 64 + n];
  }
  for (int idx = tid; idx < 128 * 64; idx += 256) {
    int n = idx >> 6, k = idx & 63;
    wa2t[idx] = (_Float16)Wa2[k * 128 + n];
  }
  __syncthreads();

  const int wave = tid >> 5, lane = tid & 31;
  const int p = blockIdx.x * 8 + wave;  // one point (= 16 edges) per wave
  _Float16* s32 = st32[wave];
  _Float16* s64 = st64[wave];

  // edge_geom -> h = relu(eg @ W1' + b1')       (BN1 pre-folded)
  gather_eg(geom, idxj, p, s32);
  v16h aEG = ld_fragA(s32, 32);
#pragma unroll
  for (int t = 0; t < 4; ++t) {
    v16h b = ld_fragB(w1t + 16 * t * 32, 32);
    v8f z = {};
    z = wmma16(aEG, b, z);
    store_rows_relu(z, s64, 64, 16 * t, ws[WS_B1HAT + 16 * t + (lane & 15)]);
  }

  // z2 = h @ W2 ; edge_feat = relu(z2*scale2 + shift2)   (BN2+b2 pre-folded)
  v8f fC[4];
#pragma unroll
  for (int t = 0; t < 4; ++t) { v8f z = {}; fC[t] = z; }
#pragma unroll
  for (int k0 = 0; k0 < 64; k0 += 32) {
    v16h aH = ld_fragA(s64 + k0, 64);
#pragma unroll
    for (int t = 0; t < 4; ++t) {
      v16h b = ld_fragB(w2t + 16 * t * 64 + k0, 64);
      fC[t] = wmma16(aH, b, fC[t]);
    }
  }
#pragma unroll
  for (int t = 0; t < 4; ++t) {
    const int col = 16 * t + (lane & 15);
    const float sc = ws[WS_SCALE2 + col], sh = ws[WS_SHIFT2 + col];
#pragma unroll
    for (int i = 0; i < 8; ++i) {
      float v = fC[t][i] * sc + sh;
      fC[t][i] = v > 0.f ? v : 0.f;
    }
  }

  // aux gating: u = relu([aux_i,aux_j] @ Wa1 + ba1); gb = u @ Wa2 + ba2
  gather_ax(aux, idxj, p, s32);  // reuse st32 (DS ops in-order per wave)
  v16h aAX = ld_fragA(s32, 32);
#pragma unroll
  for (int t = 0; t < 4; ++t) {
    v16h b = ld_fragB(wa1t + 16 * t * 32, 32);
    v8f z = {};
    z = wmma16(aAX, b, z);
    store_rows_relu(z, s64, 64, 16 * t, ba1[16 * t + (lane & 15)]);  // reuse st64
  }
  v8f g[8];
#pragma unroll
  for (int t = 0; t < 8; ++t) { v8f z = {}; g[t] = z; }
#pragma unroll
  for (int k0 = 0; k0 < 64; k0 += 32) {
    v16h aU = ld_fragA(s64 + k0, 64);
#pragma unroll
    for (int t = 0; t < 8; ++t) {
      v16h b = ld_fragB(wa2t + 16 * t * 64 + k0, 64);
      g[t] = wmma16(aU, b, g[t]);
    }
  }

  // mod = sigmoid(gb[:,:64]+1)*edge_feat + gb[:,64:], max-pool over 16 edges
  float pv[4];
#pragma unroll
  for (int t = 0; t < 4; ++t) {
    const int col = 16 * t + (lane & 15);
    const float bg = ba2[col], bb = ba2[col + 64];
    float pm = -3.4e38f;
#pragma unroll
    for (int i = 0; i < 8; ++i) {
      float gv    = g[t][i] + bg + 1.f;
      float gamma = 1.f / (1.f + __expf(-gv));
      float beta  = g[t + 4][i] + bb;
      float m     = gamma * fC[t][i] + beta;
      pm = fmaxf(pm, m);
    }
    pm = fmaxf(pm, __shfl_xor(pm, 16, 32));  // combine rows 0-7 with 8-15
    pv[t] = pm;
  }

  // layernorm over 64 features (lanes 0..15 hold 4 cols each; 16..31 duplicate)
  float s = pv[0] + pv[1] + pv[2] + pv[3];
#pragma unroll
  for (int d = 1; d < 16; d <<= 1) s += __shfl_xor(s, d, 32);
  const float mean = s * (1.f / 64.f);
  float q = 0.f;
#pragma unroll
  for (int t = 0; t < 4; ++t) { float d = pv[t] - mean; q += d * d; }
#pragma unroll
  for (int d = 1; d < 16; d <<= 1) q += __shfl_xor(q, d, 32);
  const float inv = rsqrtf(q * (1.f / 64.f) + EPSV);

  if (lane < 16) {
#pragma unroll
    for (int t = 0; t < 4; ++t) {
      const int col = 16 * t + lane;
      float v = lng[col] * (pv[t] - mean) * inv + lnb[col];
      out[(size_t)p * 64 + col] = v > 0.f ? v : 0.f;
    }
  }
}

// ---------------------------------------------------------------- launcher

extern "C" void kernel_launch(void* const* d_in, const int* in_sizes, int n_in,
                              void* d_out, int out_size, void* d_ws,
                              size_t ws_size, hipStream_t stream) {
  const float* geom  = (const float*)d_in[0];
  const float* aux   = (const float*)d_in[1];
  const float* W1    = (const float*)d_in[2];
  const float* b1    = (const float*)d_in[3];
  const float* bn1_g = (const float*)d_in[4];
  const float* bn1_b = (const float*)d_in[5];
  const float* W2    = (const float*)d_in[6];
  // b2 (d_in[7]) folds out analytically in BN2
  const float* bn2_g = (const float*)d_in[8];
  const float* bn2_b = (const float*)d_in[9];
  const float* Wa1   = (const float*)d_in[10];
  const float* ba1   = (const float*)d_in[11];
  const float* Wa2   = (const float*)d_in[12];
  const float* ba2   = (const float*)d_in[13];
  const float* lng   = (const float*)d_in[14];
  const float* lnb   = (const float*)d_in[15];
  const int*   idxj  = ((const int*)d_in[17]) + E_TOT;  // edge_index row 1
  float* out = (float*)d_out;
  float* ws  = (float*)d_ws;

  hipMemsetAsync(d_ws, 0, WS_TOTAL * sizeof(float), stream);
  s1_cov_kernel<<<256, 256, 0, stream>>>(geom, idxj, ws + WS_S1);
  fold_bn1_kernel<<<1, 64, 0, stream>>>(W1, b1, bn1_g, bn1_b, ws);
  s2_stats_kernel<<<2048, 256, 0, stream>>>(geom, idxj, W2, ws);
  fold_bn2_kernel<<<1, 64, 0, stream>>>(bn2_g, bn2_b, ws);
  s3_main_kernel<<<P_PTS / 8, 256, 0, stream>>>(geom, aux, idxj, W2, Wa1, ba1,
                                                Wa2, ba2, lng, lnb, ws, out);
}
